// HierarchicalFusionBlock_without_self_attention_30159260352669
// MI455X (gfx1250) — matrix operand order repaired and checked by reading, not compiled, not run on hardware
//
#include <hip/hip_runtime.h>
#include <hip/hip_bf16.h>

// ---------------------------------------------------------------------------
// Problem constants (from the reference)
// ---------------------------------------------------------------------------
constexpr int CB = 4, CL = 1024, CH = 768, CNH = 12;
constexpr int CNPG = 512, CNMAX = 512;
constexpr int CN = CB * CNPG;          // 2048 graph nodes
constexpr int CE = CB * CNPG * 8;      // 16384 edges
constexpr int CI = 1536, CS = 16, CR = 48, CK = 4;
constexpr int CD = CH / CNH;           // 64 head dim

typedef __attribute__((ext_vector_type(16))) __bf16 v16bf;
typedef __attribute__((ext_vector_type(8)))  __bf16 v8bf;
typedef __attribute__((ext_vector_type(4)))  __bf16 v4bf;
typedef __attribute__((ext_vector_type(8)))  float  v8f;

__device__ __forceinline__ __bf16 f2bf(float f) {
  unsigned u = __builtin_bit_cast(unsigned, f);
  unsigned r = u + 0x7FFFu + ((u >> 16) & 1u);   // round-to-nearest-even
  unsigned short h = (unsigned short)(r >> 16);
  return __builtin_bit_cast(__bf16, h);
}
// two contiguous ds_load_b128 -> one 16-elem bf16 fragment
__device__ __forceinline__ v16bf ld16(const __bf16* p) {
  v8bf lo = *(const v8bf*)p;
  v8bf hi = *(const v8bf*)(p + 8);
  return __builtin_shufflevector(lo, hi, 0, 1, 2, 3, 4, 5, 6, 7,
                                 8, 9, 10, 11, 12, 13, 14, 15);
}
// pack 4 floats -> 4 bf16, one 8-byte LDS store
__device__ __forceinline__ void st4bf(__bf16* p, float4 v) {
  v4bf t; t[0] = f2bf(v.x); t[1] = f2bf(v.y); t[2] = f2bf(v.z); t[3] = f2bf(v.w);
  *(v4bf*)p = t;
}
// column permutation so an A fragment is contiguous per half-wave:
// positions [0..15] hold K {0..7,16..23}; [16..31] hold K {8..15,24..31}
__device__ __forceinline__ int posc(int c) {
  return (c < 8) ? c : (c < 16) ? c + 8 : (c < 24) ? c - 8 : c;
}

// ---------------------------------------------------------------------------
// Generic WMMA GEMM:  C[M,N] = act( concat-A[M,K] @ B[K,N] + bias + resid )
// Block 256 threads (8 waves), tile 128x64; each wave: 16x64 (1 A-frag,
// 4 B-frags, 4 WMMAs per 32-K step). bf16 operands, f32 accumulate.
// Fast staging path (float4 loads, v4bf stores) when the 32-wide K-slice is
// contained in one A segment; elementwise fallback otherwise.
// ---------------------------------------------------------------------------
struct SegDesc { const float* ptr; const int* idx; int width; int stride; };

__global__ __launch_bounds__(256)
void gemm_wmma(SegDesc s0, SegDesc s1, SegDesc s2,
               const float* __restrict__ Bmat,
               const float* __restrict__ bias,
               const float* __restrict__ resid,
               float* __restrict__ Cout,
               int M, int N, int Kdim, int act) {
  __shared__ __bf16 As[128][40];   // k-contiguous columns, padded stride (80B)
  __shared__ __bf16 Bt[64][40];    // transposed B tile: Bt[n][k]

  const int tid  = threadIdx.x;
  const int lane = tid & 31;
  const int wave = tid >> 5;
  const int half = lane >> 4;
  const int l15  = lane & 15;
  const int m0 = blockIdx.y * 128;
  const int n0 = blockIdx.x * 64;

  v8f acc[4] = {};
  const int st1 = s0.width;
  const int st2 = s0.width + s1.width;

  for (int k0 = 0; k0 < Kdim; k0 += 32) {
    // ---- uniform segment selection for this K-slice (no exec divergence) ---
    const float* sp = nullptr; const int* sidx = nullptr; int sstride = 0, soff = 0;
    if (k0 + 32 <= st1)                      { sp = s0.ptr; sidx = s0.idx; sstride = s0.stride; soff = k0; }
    else if (k0 >= st1 && k0 + 32 <= st2)    { sp = s1.ptr; sidx = s1.idx; sstride = s1.stride; soff = k0 - st1; }
    else if (k0 >= st2 && k0 + 32 <= Kdim)   { sp = s2.ptr; sidx = s2.idx; sstride = s2.stride; soff = k0 - st2; }

    if (sp) {
      // fast: 4 floats per thread-iteration, vector loads + 8B LDS stores
      for (int i = tid; i < 128 * 8; i += 256) {
        int r = i >> 3, c4 = (i & 7) * 4;
        int gr = m0 + r;
        float4 v = make_float4(0.f, 0.f, 0.f, 0.f);
        if (gr < M) {
          int rr = sidx ? sidx[gr] : gr;
          v = *(const float4*)(sp + (long long)rr * sstride + soff + c4);
        }
        st4bf(&As[r][c4], v);
      }
    } else {
      // slow: per-element with segment branches (ragged concat / K tail)
      for (int i = tid; i < 128 * 32; i += 256) {
        int r = i >> 5, c = i & 31;
        int gr = m0 + r, gk = k0 + c;
        float v = 0.f;
        if (gr < M) {
          if (gk < st1) {
            int rr = s0.idx ? s0.idx[gr] : gr;
            v = s0.ptr[(long long)rr * s0.stride + gk];
          } else if (gk < st2) {
            int rr = s1.idx ? s1.idx[gr] : gr;
            v = s1.ptr[(long long)rr * s1.stride + (gk - st1)];
          } else if (gk < Kdim) {
            int rr = s2.idx ? s2.idx[gr] : gr;
            v = s2.ptr[(long long)rr * s2.stride + (gk - st2)];
          }
        }
        As[r][c] = f2bf(v);
      }
    }

    // ---- B tile (32 x 64), transposed into Bt[n][k] ------------------------
    if (k0 + 32 <= Kdim && n0 + 64 <= N) {
      for (int i = tid; i < 32 * 16; i += 256) {
        int r = i >> 4, c4 = (i & 15) * 4;
        float4 v = *(const float4*)(Bmat + (long long)(k0 + r) * N + n0 + c4);
        Bt[c4 + 0][r] = f2bf(v.x);
        Bt[c4 + 1][r] = f2bf(v.y);
        Bt[c4 + 2][r] = f2bf(v.z);
        Bt[c4 + 3][r] = f2bf(v.w);
      }
    } else {
      for (int i = tid; i < 32 * 64; i += 256) {
        int r = i >> 6, c = i & 63;
        int gk = k0 + r, gn = n0 + c;
        float v = (gk < Kdim && gn < N) ? Bmat[(long long)gk * N + gn] : 0.f;
        Bt[c][r] = f2bf(v);
      }
    }
    __syncthreads();

    v16bf a = ld16(&As[wave * 16 + l15][half * 16]);
#pragma unroll
    for (int nt = 0; nt < 4; ++nt) {
      v16bf b = ld16(&Bt[nt * 16 + l15][half * 16]);
      acc[nt] = __builtin_amdgcn_wmma_f32_16x16x32_bf16(false, a, false, b,
                                                        (short)0, acc[nt],
                                                        false, false);
    }
    __syncthreads();
  }

  // epilogue: C layout M = r + 8*half, N = l15 (per 16-col subtile)
#pragma unroll
  for (int nt = 0; nt < 4; ++nt) {
#pragma unroll
    for (int r = 0; r < 8; ++r) {
      int row = m0 + wave * 16 + r + 8 * half;
      int col = n0 + nt * 16 + l15;
      if (row >= M || col >= N) continue;
      float v = acc[nt][r];
      if (bias)  v += bias[col];
      if (resid) v += resid[(long long)row * N + col];
      if (act == 1) v = fmaxf(v, 0.f);
      else if (act == 2) v = (v > 20.f) ? v : log1pf(__expf(v));
      Cout[(long long)row * N + col] = v;
    }
  }
}

// ---------------------------------------------------------------------------
// Flash-attention: block = (batch, head, 64 queries), 4 waves; each wave owns
// a 16-query tile. K/V tiles (32 keys) shared in LDS in fragment-friendly
// layouts; online softmax with 16-lane butterfly reductions.
// ---------------------------------------------------------------------------
__global__ __launch_bounds__(128)
void attn_wmma(const float* __restrict__ Q, const float* __restrict__ Km,
               const float* __restrict__ Vm, const int* __restrict__ kmask,
               float* __restrict__ Ctx, int Lq, int Lk) {
  __shared__ __bf16 Ksh[32][72];        // [key][dim] : contig contraction for QK^T
  __shared__ __bf16 Vt[64][40];         // [dim][key] : contig contraction for P*V
  __shared__ __bf16 Psh[4][16][40];     // per-wave P scratch, key-contiguous cols
  __shared__ float  Mv[32];

  const int tid  = threadIdx.x;
  const int lane = tid & 31;
  const int wave = tid >> 5;
  const int half = lane >> 4;
  const int l15  = lane & 15;
  const int b    = blockIdx.z;
  const int head = blockIdx.y;
  const int q0   = blockIdx.x * 64 + wave * 16;

  // preload Q fragments (2 chunks of 32 over d=64), pre-scaled by 1/sqrt(d)
  v16bf qa[2];
  {
    const int qrow = q0 + l15;
    const float* qp = Q + ((long long)(b * Lq + qrow)) * CH + head * CD;
    const int kb = half * 16;
#pragma unroll
    for (int c = 0; c < 2; ++c)
#pragma unroll
      for (int i = 0; i < 8; ++i) {
        qa[c][i]     = f2bf(qp[c * 32 + kb + i] * 0.125f);
        qa[c][8 + i] = f2bf(qp[c * 32 + kb + 8 + i] * 0.125f);
      }
  }

  float mrow[8], lrow[8];
  v8f o[4] = {};
#pragma unroll
  for (int r = 0; r < 8; ++r) { mrow[r] = -1e30f; lrow[r] = 0.f; }

  for (int j = 0; j < Lk; j += 32) {
    __syncthreads();
    // stage K (row layout, vectorized) and V (transposed)
    for (int i = tid; i < 32 * 16; i += 128) {
      int key = i >> 4, d4 = (i & 15) * 4;
      long long srcb = ((long long)(b * Lk + j + key)) * CH + head * CD + d4;
      float4 kv = *(const float4*)(Km + srcb);
      st4bf(&Ksh[key][d4], kv);
      float4 vv = *(const float4*)(Vm + srcb);
      Vt[d4 + 0][key] = f2bf(vv.x);
      Vt[d4 + 1][key] = f2bf(vv.y);
      Vt[d4 + 2][key] = f2bf(vv.z);
      Vt[d4 + 3][key] = f2bf(vv.w);
    }
    if (tid < 32) Mv[tid] = (kmask[b * Lk + j + tid] > 0) ? 0.f : -1e9f;
    __syncthreads();

    // S = Q K^T  (two 16-key tiles, contraction d=64 in two chunks)
    v8f s[2];
#pragma unroll
    for (int nt = 0; nt < 2; ++nt) {
      v8f accs = {};
#pragma unroll
      for (int kc = 0; kc < 2; ++kc) {
        v16bf bfr = ld16(&Ksh[nt * 16 + l15][kc * 32 + half * 16]);
        accs = __builtin_amdgcn_wmma_f32_16x16x32_bf16(false, qa[kc], false, bfr,
                                                       (short)0, accs, false, false);
      }
      float mv = Mv[nt * 16 + l15];
#pragma unroll
      for (int r = 0; r < 8; ++r) accs[r] += mv;
      s[nt] = accs;
    }

    // online softmax (row = r + 8*half; the 16 N-lanes hold the key axis)
    float alpha[8];
#pragma unroll
    for (int r = 0; r < 8; ++r) {
      float mx = fmaxf(s[0][r], s[1][r]);
#pragma unroll
      for (int off = 8; off >= 1; off >>= 1)
        mx = fmaxf(mx, __shfl_xor(mx, off, 32));
      float mnew = fmaxf(mrow[r], mx);
      float a0 = __expf(mrow[r] - mnew);
      float p0 = __expf(s[0][r] - mnew);
      float p1 = __expf(s[1][r] - mnew);
      float rs = p0 + p1;
#pragma unroll
      for (int off = 8; off >= 1; off >>= 1)
        rs += __shfl_xor(rs, off, 32);
      lrow[r] = lrow[r] * a0 + rs;
      mrow[r] = mnew;
      alpha[r] = a0;
      s[0][r] = p0; s[1][r] = p1;
    }
#pragma unroll
    for (int nt = 0; nt < 4; ++nt)
#pragma unroll
      for (int r = 0; r < 8; ++r) o[nt][r] *= alpha[r];

    // stage P (C layout -> row-major A layout) then accumulate O += P V
#pragma unroll
    for (int nt = 0; nt < 2; ++nt)
#pragma unroll
      for (int r = 0; r < 8; ++r)
        Psh[wave][r + 8 * half][nt * 16 + l15] = f2bf(s[nt][r]);
    __syncthreads();

    v16bf pa = ld16(&Psh[wave][l15][half * 16]);
#pragma unroll
    for (int nt = 0; nt < 4; ++nt) {
      v16bf bfr = ld16(&Vt[nt * 16 + l15][half * 16]);
      o[nt] = __builtin_amdgcn_wmma_f32_16x16x32_bf16(false, pa, false, bfr,
                                                      (short)0, o[nt], false, false);
    }
  }

#pragma unroll
  for (int nt = 0; nt < 4; ++nt)
#pragma unroll
    for (int r = 0; r < 8; ++r) {
      int row = q0 + r + 8 * half;
      float v = o[nt][r] / lrow[r];
      Ctx[((long long)(b * Lq + row)) * CH + head * CD + nt * 16 + l15] = v;
    }
}

// ---------------------------------------------------------------------------
// Elementwise / reduction helper kernels
// ---------------------------------------------------------------------------
__global__ __launch_bounds__(256)
void k_layernorm(float* __restrict__ X, const float* __restrict__ g,
                 const float* __restrict__ bb, int Hd) {
  __shared__ float r1[256], r2[256];
  const int tid = threadIdx.x;
  float* x = X + (long long)blockIdx.x * Hd;
  float v[4]; int cnt = 0;
  float s = 0.f, sq = 0.f;
  for (int c = tid; c < Hd; c += 256) { float t = x[c]; v[cnt++] = t; s += t; sq += t * t; }
  r1[tid] = s; r2[tid] = sq; __syncthreads();
  for (int o = 128; o > 0; o >>= 1) {
    if (tid < o) { r1[tid] += r1[tid + o]; r2[tid] += r2[tid + o]; }
    __syncthreads();
  }
  float mean = r1[0] / Hd;
  float var  = r2[0] / Hd - mean * mean;
  float inv  = rsqrtf(var + 1e-12f);
  cnt = 0;
  for (int c = tid; c < Hd; c += 256)
    x[c] = (v[cnt++] - mean) * inv * g[c] + bb[c];
}

__global__ __launch_bounds__(256)
void k_rmsnorm(const float* __restrict__ X, const float* __restrict__ res,
               const int* __restrict__ mask, const float* __restrict__ w,
               float* __restrict__ Y, int Hd, float eps) {
  __shared__ float r1[256];
  const int tid = threadIdx.x;
  const long long row = blockIdx.x;
  const float* x = X + row * Hd;
  float v[4]; int cnt = 0;
  float sq = 0.f;
  for (int c = tid; c < Hd; c += 256) {
    float t = x[c];
    if (res) t += res[row * Hd + c];
    v[cnt++] = t; sq += t * t;
  }
  r1[tid] = sq; __syncthreads();
  for (int o = 128; o > 0; o >>= 1) {
    if (tid < o) r1[tid] += r1[tid + o];
    __syncthreads();
  }
  float inv = rsqrtf(r1[0] / Hd + eps);
  float mm = 1.f;
  if (mask) mm = (mask[row] > 0) ? 1.f : 0.f;
  cnt = 0;
  for (int c = tid; c < Hd; c += 256)
    Y[row * Hd + c] = v[cnt++] * inv * w[c] * mm;
}

__global__ void k_counts(const int* __restrict__ batch, int* counts, int n) {
  int i = blockIdx.x * blockDim.x + threadIdx.x;
  if (i < n) atomicAdd(&counts[batch[i]], 1);
}
__global__ void k_starts(const int* __restrict__ counts, int* starts) {
  if (threadIdx.x == 0) {
    int acc = 0;
    for (int b = 0; b < CB; ++b) { starts[b] = acc; acc += counts[b]; }
  }
}
__global__ void k_pos(const int* __restrict__ batch, const int* __restrict__ starts,
                      int* __restrict__ pos, int n) {
  int i = blockIdx.x * blockDim.x + threadIdx.x;
  if (i < n) pos[i] = i - starts[batch[i]];
}
__global__ void k_scatter(const float* __restrict__ x, const int* __restrict__ batch,
                          const int* __restrict__ pos, float* __restrict__ ghs, int total) {
  int i = blockIdx.x * blockDim.x + threadIdx.x;
  if (i >= total) return;
  int node = i / CH, c = i % CH;
  ghs[((long long)batch[node] * CNMAX + pos[node]) * CH + c] = x[(long long)node * CH + c];
}
__global__ void k_gmask(const int* __restrict__ counts, int* __restrict__ gmask) {
  int i = blockIdx.x * blockDim.x + threadIdx.x;
  if (i >= CB * CNMAX) return;
  gmask[i] = ((i % CNMAX) < counts[i / CNMAX]) ? 1 : 0;
}
__global__ void k_gather(const float* __restrict__ g, const int* __restrict__ batch,
                         const int* __restrict__ pos, float* __restrict__ x, int total) {
  int i = blockIdx.x * blockDim.x + threadIdx.x;
  if (i >= total) return;
  int node = i / CH, c = i % CH;
  x[(long long)node * CH + c] = g[((long long)batch[node] * CNMAX + pos[node]) * CH + c];
}
__global__ void k_segsum(const float* __restrict__ src, const int* __restrict__ seg,
                         float* __restrict__ out, int rows, int cols) {
  long long i = (long long)blockIdx.x * blockDim.x + threadIdx.x;
  if (i >= (long long)rows * cols) return;
  int r = (int)(i / cols), c = (int)(i % cols);
  atomicAdd(&out[(long long)seg[r] * cols + c], src[i]);
}
__global__ void k_conv_silu(const float* __restrict__ xz, const int* __restrict__ mask,
                            const float* __restrict__ cw, const float* __restrict__ cb,
                            float* __restrict__ xc) {
  long long idx = (long long)blockIdx.x * blockDim.x + threadIdx.x;
  if (idx >= (long long)CB * CL * CI) return;
  int i = (int)(idx % CI);
  int t = (int)((idx / CI) % CL);
  int b = (int)(idx / ((long long)CI * CL));
  float acc = cb[i];
#pragma unroll
  for (int k = 0; k < CK; ++k) {
    int tt = t - (CK - 1) + k;
    if (tt >= 0)
      acc += xz[((long long)(b * CL + tt)) * (2 * CI) + i] * cw[i * CK + k];
  }
  float s = acc / (1.f + __expf(-acc));            // silu
  float m = (mask[b * CL + t] > 0) ? 1.f : 0.f;
  xc[((long long)(b * CL + t)) * CI + i] = s * m;
}
__global__ void k_scan(const float* __restrict__ dt, const float* __restrict__ xc,
                       const float* __restrict__ dbc, const float* __restrict__ A_log,
                       const float* __restrict__ Dp, const float* __restrict__ xz,
                       float* __restrict__ Y) {
  int idx = blockIdx.x * blockDim.x + threadIdx.x;
  if (idx >= CB * CI) return;
  int b = idx / CI, i = idx % CI;
  float A[CS], st[CS];
#pragma unroll
  for (int s = 0; s < CS; ++s) { A[s] = -__expf(A_log[i * CS + s]); st[s] = 0.f; }
  for (int t = 0; t < CL; ++t) {
    long long o = (long long)(b * CL + t);
    float dtv = dt[o * CI + i];
    float xv  = xc[o * CI + i];
    const float* Bc = dbc + o * (CR + 2 * CS) + CR;
    const float* Cc = Bc + CS;
    float y = 0.f;
#pragma unroll
    for (int s = 0; s < CS; ++s) {
      st[s] = st[s] * __expf(dtv * A[s]) + dtv * xv * Bc[s];
      y += st[s] * Cc[s];
    }
    y += xv * Dp[i];
    float z = xz[o * (2 * CI) + CI + i];
    y *= z / (1.f + __expf(-z));
    Y[o * CI + i] = y;
  }
}
__global__ void k_add_cls(float* __restrict__ h2, const float* __restrict__ upd) {
  int i = blockIdx.x * blockDim.x + threadIdx.x;
  if (i >= CB * CH) return;
  int b = i / CH, c = i % CH;
  h2[(long long)b * CL * CH + c] += upd[i];
}

// ---------------------------------------------------------------------------
// Host-side orchestration
// ---------------------------------------------------------------------------
static inline SegDesc seg(const float* p, const int* idx, int w, int s) {
  SegDesc d; d.ptr = p; d.idx = idx; d.width = w; d.stride = s; return d;
}
static inline void launch_gemm(hipStream_t st, SegDesc s0, SegDesc s1, SegDesc s2,
                               const float* Bm, const float* bias, const float* resid,
                               float* out, int M, int N, int Kd, int act) {
  dim3 g((N + 63) / 64, (M + 127) / 128);
  gemm_wmma<<<g, 256, 0, st>>>(s0, s1, s2, Bm, bias, resid, out, M, N, Kd, act);
}

extern "C" void kernel_launch(void* const* d_in, const int* in_sizes, int n_in,
                              void* d_out, int out_size, void* d_ws, size_t ws_size,
                              hipStream_t stream) {
  // ---- inputs --------------------------------------------------------------
  const float* hidden = (const float*)d_in[0];           // B,L,H
  const int*   amask  = (const int*)d_in[1];             // B,L
  const float* gx     = (const float*)d_in[2];           // N,9
  const int*   eidx   = (const int*)d_in[3];             // 2,E
  const float* eattr  = (const float*)d_in[4];           // E,3
  const int*   gbatch = (const int*)d_in[5];             // N
  const int* src = eidx, *dst = eidx + CE;
  // params, flattened in dict order
  int p = 6;
  const float* sWq = (const float*)d_in[p++]; const float* sWk = (const float*)d_in[p++];
  const float* sWv = (const float*)d_in[p++]; const float* sWo = (const float*)d_in[p++];
  const float* sbq = (const float*)d_in[p++]; const float* sbk = (const float*)d_in[p++];
  const float* sbv = (const float*)d_in[p++]; const float* sbo = (const float*)d_in[p++];
  const float* slg = (const float*)d_in[p++]; const float* slb = (const float*)d_in[p++];
  const float* gWq = (const float*)d_in[p++]; const float* gWk = (const float*)d_in[p++];
  const float* gWv = (const float*)d_in[p++]; const float* gWo = (const float*)d_in[p++];
  const float* gbq = (const float*)d_in[p++]; const float* gbk = (const float*)d_in[p++];
  const float* gbv = (const float*)d_in[p++]; const float* gbo = (const float*)d_in[p++];
  const float* glg = (const float*)d_in[p++]; const float* glb = (const float*)d_in[p++];
  const float* We0 = (const float*)d_in[p++]; const float* be0 = (const float*)d_in[p++];
  const float* Wn0 = (const float*)d_in[p++]; const float* bn0 = (const float*)d_in[p++];
  const float* We1 = (const float*)d_in[p++]; const float* be1 = (const float*)d_in[p++];
  const float* Wn1 = (const float*)d_in[p++]; const float* bn1 = (const float*)d_in[p++];
  const float* We2 = (const float*)d_in[p++]; const float* be2 = (const float*)d_in[p++];
  const float* Wn2 = (const float*)d_in[p++]; const float* bn2 = (const float*)d_in[p++];
  const float* mnw  = (const float*)d_in[p++]; const float* minp = (const float*)d_in[p++];
  const float* mcw  = (const float*)d_in[p++]; const float* mcb  = (const float*)d_in[p++];
  const float* mxp  = (const float*)d_in[p++]; const float* mdtp = (const float*)d_in[p++];
  const float* mdtb = (const float*)d_in[p++]; const float* mAlg = (const float*)d_in[p++];
  const float* mD   = (const float*)d_in[p++]; const float* mop  = (const float*)d_in[p++];
  const float* Wm = (const float*)d_in[p++]; const float* bm = (const float*)d_in[p++];
  const float* fnw = (const float*)d_in[p++];

  // ---- output layout -------------------------------------------------------
  float* outp  = (float*)d_out;
  float* out_h  = outp;                                  // B*L*H
  float* out_x  = outp + (long long)CB * CL * CH;        // N*H
  float* out_ea = out_x + (long long)CN * CH;            // E*H

  // ---- workspace carving ---------------------------------------------------
  float* w = (float*)d_ws;
  size_t cur = 0;
  auto alloc = [&](size_t n) { float* q = w + cur; cur += n; return q; };
  float* x0   = alloc((size_t)CN * CH);
  float* mA   = alloc((size_t)CE * CH);
  float* agg  = alloc((size_t)CN * CH);
  float* ghs  = alloc((size_t)CB * CNMAX * CH);
  float* q1   = alloc((size_t)CB * CL * CH);
  float* k1   = alloc((size_t)CB * CNMAX * CH);
  float* v1   = alloc((size_t)CB * CNMAX * CH);
  float* ctx1 = alloc((size_t)CB * CL * CH);
  float* h1   = alloc((size_t)CB * CL * CH);
  float* q2   = alloc((size_t)CB * CNMAX * CH);
  float* k2   = alloc((size_t)CB * CL * CH);
  float* v2   = alloc((size_t)CB * CL * CH);
  float* ctx2 = alloc((size_t)CB * CNMAX * CH);
  float* gbuf = alloc((size_t)CB * CNMAX * CH);
  float* x1   = alloc((size_t)CN * CH);
  float* mB   = alloc((size_t)CE * CH);
  float* x2b  = alloc((size_t)CN * CH);
  float* hn   = alloc((size_t)CB * CL * CH);
  float* xz   = alloc((size_t)CB * CL * 2 * CI);
  float* xc   = alloc((size_t)CB * CL * CI);
  float* dbc  = alloc((size_t)CB * CL * (CR + 2 * CS));
  float* dtb  = alloc((size_t)CB * CL * CI);
  float* yb   = alloc((size_t)CB * CL * CI);
  float* h2   = alloc((size_t)CB * CL * CH);
  float* gemb = alloc((size_t)CB * CH);
  float* updb = alloc((size_t)CB * CH);
  int*   counts = (int*)alloc(64);
  int*   starts = counts + CB;
  int*   posb   = (int*)alloc(CN);
  int*   gmask  = (int*)alloc((size_t)CB * CNMAX);
  (void)ws_size; (void)n_in; (void)in_sizes; (void)out_size;

  const SegDesc nil = seg(nullptr, nullptr, 0, 0);

  // ---- graph2batch bookkeeping --------------------------------------------
  hipMemsetAsync(counts, 0, CB * sizeof(int), stream);
  k_counts<<<(CN + 255) / 256, 256, 0, stream>>>(gbatch, counts, CN);
  k_starts<<<1, 1, 0, stream>>>(counts, starts);
  k_pos<<<(CN + 255) / 256, 256, 0, stream>>>(gbatch, starts, posb, CN);
  k_gmask<<<(CB * CNMAX + 255) / 256, 256, 0, stream>>>(counts, gmask);

  // ---- MPNN g0 -------------------------------------------------------------
  launch_gemm(stream, seg(gx, src, 9, 9), seg(gx, dst, 9, 9), seg(eattr, nullptr, 3, 3),
              We0, be0, nullptr, mA, CE, CH, 21, 1);
  hipMemsetAsync(agg, 0, (size_t)CN * CH * sizeof(float), stream);
  k_segsum<<<(int)(((long long)CE * CH + 255) / 256), 256, 0, stream>>>(mA, dst, agg, CE, CH);
  launch_gemm(stream, seg(gx, nullptr, 9, 9), seg(agg, nullptr, CH, CH), nil,
              Wn0, bn0, nullptr, x0, CN, CH, 9 + CH, 1);

  // ---- graph2batch scatter -------------------------------------------------
  hipMemsetAsync(ghs, 0, (size_t)CB * CNMAX * CH * sizeof(float), stream);
  k_scatter<<<(int)(((long long)CN * CH + 255) / 256), 256, 0, stream>>>(x0, gbatch, posb, ghs, CN * CH);

  // ---- seq cross-attention (q = hidden, kv = ghs) --------------------------
  launch_gemm(stream, seg(hidden, nullptr, CH, CH), nil, nil, sWq, sbq, nullptr, q1, CB * CL, CH, CH, 0);
  launch_gemm(stream, seg(ghs, nullptr, CH, CH), nil, nil, sWk, sbk, nullptr, k1, CB * CNMAX, CH, CH, 0);
  launch_gemm(stream, seg(ghs, nullptr, CH, CH), nil, nil, sWv, sbv, nullptr, v1, CB * CNMAX, CH, CH, 0);
  attn_wmma<<<dim3(CL / 64, CNH, CB), 128, 0, stream>>>(q1, k1, v1, gmask, ctx1, CL, CNMAX);
  launch_gemm(stream, seg(ctx1, nullptr, CH, CH), nil, nil, sWo, sbo, hidden, h1, CB * CL, CH, CH, 0);
  k_layernorm<<<CB * CL, 256, 0, stream>>>(h1, slg, slb, CH);

  // ---- graph cross-attention (q = ghs, kv = hidden) ------------------------
  launch_gemm(stream, seg(ghs, nullptr, CH, CH), nil, nil, gWq, gbq, nullptr, q2, CB * CNMAX, CH, CH, 0);
  launch_gemm(stream, seg(hidden, nullptr, CH, CH), nil, nil, gWk, gbk, nullptr, k2, CB * CL, CH, CH, 0);
  launch_gemm(stream, seg(hidden, nullptr, CH, CH), nil, nil, gWv, gbv, nullptr, v2, CB * CL, CH, CH, 0);
  attn_wmma<<<dim3(CNMAX / 64, CNH, CB), 128, 0, stream>>>(q2, k2, v2, amask, ctx2, CNMAX, CL);
  launch_gemm(stream, seg(ctx2, nullptr, CH, CH), nil, nil, gWo, gbo, ghs, gbuf, CB * CNMAX, CH, CH, 0);
  k_layernorm<<<CB * CNMAX, 256, 0, stream>>>(gbuf, glg, glb, CH);

  // ---- back to node space, MPNN g1 / g2 ------------------------------------
  k_gather<<<(int)(((long long)CN * CH + 255) / 256), 256, 0, stream>>>(gbuf, gbatch, posb, x1, CN * CH);
  launch_gemm(stream, seg(x1, src, CH, CH), seg(x1, dst, CH, CH), seg(mA, nullptr, CH, CH),
              We1, be1, nullptr, mB, CE, CH, 3 * CH, 1);
  hipMemsetAsync(agg, 0, (size_t)CN * CH * sizeof(float), stream);
  k_segsum<<<(int)(((long long)CE * CH + 255) / 256), 256, 0, stream>>>(mB, dst, agg, CE, CH);
  launch_gemm(stream, seg(x1, nullptr, CH, CH), seg(agg, nullptr, CH, CH), nil,
              Wn1, bn1, nullptr, x2b, CN, CH, 2 * CH, 1);

  launch_gemm(stream, seg(x2b, src, CH, CH), seg(x2b, dst, CH, CH), seg(mB, nullptr, CH, CH),
              We2, be2, nullptr, out_ea, CE, CH, 3 * CH, 1);
  hipMemsetAsync(agg, 0, (size_t)CN * CH * sizeof(float), stream);
  k_segsum<<<(int)(((long long)CE * CH + 255) / 256), 256, 0, stream>>>(out_ea, dst, agg, CE, CH);
  launch_gemm(stream, seg(x2b, nullptr, CH, CH), seg(agg, nullptr, CH, CH), nil,
              Wn2, bn2, nullptr, out_x, CN, CH, 2 * CH, 1);

  // ---- Mamba block on h1 ---------------------------------------------------
  k_rmsnorm<<<CB * CL, 256, 0, stream>>>(h1, nullptr, amask, mnw, hn, CH, 1e-6f);
  launch_gemm(stream, seg(hn, nullptr, CH, CH), nil, nil, minp, nullptr, nullptr,
              xz, CB * CL, 2 * CI, CH, 0);
  k_conv_silu<<<(int)(((long long)CB * CL * CI + 255) / 256), 256, 0, stream>>>(xz, amask, mcw, mcb, xc);
  launch_gemm(stream, seg(xc, nullptr, CI, CI), nil, nil, mxp, nullptr, nullptr,
              dbc, CB * CL, CR + 2 * CS, CI, 0);
  launch_gemm(stream, seg(dbc, nullptr, CR, CR + 2 * CS), nil, nil, mdtp, mdtb, nullptr,
              dtb, CB * CL, CI, CR, 2);
  k_scan<<<(CB * CI + 255) / 256, 256, 0, stream>>>(dtb, xc, dbc, mAlg, mD, xz, yb);
  launch_gemm(stream, seg(yb, nullptr, CI, CI), nil, nil, mop, nullptr, h1,
              h2, CB * CL, CH, CI, 0);

  // ---- fusion head ---------------------------------------------------------
  hipMemsetAsync(gemb, 0, (size_t)CB * CH * sizeof(float), stream);
  k_segsum<<<(int)(((long long)CN * CH + 255) / 256), 256, 0, stream>>>(out_x, gbatch, gemb, CN, CH);
  // upd = concat(graph_emb, cls=h2[:,0,:]) @ Wm + bm   (cls via stride L*H)
  launch_gemm(stream, seg(gemb, nullptr, CH, CH), seg(h2, nullptr, CH, CL * CH), nil,
              Wm, bm, nullptr, updb, CB, CH, 2 * CH, 0);
  k_add_cls<<<(CB * CH + 255) / 256, 256, 0, stream>>>(h2, updb);

  // ---- final residual + rmsnorm -> output h --------------------------------
  k_rmsnorm<<<CB * CL, 256, 0, stream>>>(h2, hidden, nullptr, fnw, out_h, CH, 1e-6f);
}
